// CFC_15616501088830
// MI455X (gfx1250) — matrix operand order repaired and checked by reading, not compiled, not run on hardware
//
#include <hip/hip_runtime.h>
#include <hip/hip_bf16.h>

#define HID 128
#define EIN 64
#define NNODES 20000
#define NEDGES 640000

typedef __attribute__((ext_vector_type(16))) __bf16 bf16x16;
typedef __attribute__((ext_vector_type(8)))  __bf16 bf16x8;
typedef __attribute__((ext_vector_type(8)))  float  f32x8;

// ShiftedSoftplus: softplus(x) - ln2 == max(x,0) + log(0.5 + 0.5*exp(-|x|))
// (fma folds the +1 and the -ln2 shift; exact and numerically stable)
__device__ __forceinline__ float ssp(float x) {
    float e = __expf(-fabsf(x));              // -|x| is a free src modifier
    return fmaxf(x, 0.0f) + __logf(fmaf(e, 0.5f, 0.5f));
}

__device__ __forceinline__ void atomAddF(float* p, float v) {
    __hip_atomic_fetch_add(p, v, __ATOMIC_RELAXED, __HIP_MEMORY_SCOPE_AGENT);
}

// Build a 16x32 bf16 A-operand from row-major f32 (row pointer p, chunk base kb).
// ISA layout: lane<16 -> K = kb..kb+7 (v0-3) and kb+16..kb+23 (v4-7); lanes>=16
// pass kb already offset by +8.
__device__ __forceinline__ bf16x16 pack_a_f32(const float* __restrict__ p, int kb) {
    float4 f0 = *(const float4*)(p + kb);
    float4 f1 = *(const float4*)(p + kb + 4);
    float4 f2 = *(const float4*)(p + kb + 16);
    float4 f3 = *(const float4*)(p + kb + 20);
    bf16x16 v;
    v[0]  = (__bf16)f0.x; v[1]  = (__bf16)f0.y; v[2]  = (__bf16)f0.z; v[3]  = (__bf16)f0.w;
    v[4]  = (__bf16)f1.x; v[5]  = (__bf16)f1.y; v[6]  = (__bf16)f1.z; v[7]  = (__bf16)f1.w;
    v[8]  = (__bf16)f2.x; v[9]  = (__bf16)f2.y; v[10] = (__bf16)f2.z; v[11] = (__bf16)f2.w;
    v[12] = (__bf16)f3.x; v[13] = (__bf16)f3.y; v[14] = (__bf16)f3.z; v[15] = (__bf16)f3.w;
    return v;
}

__device__ __forceinline__ f32x8 wmma_bf16(bf16x16 a, bf16x16 b, f32x8 c) {
    return __builtin_amdgcn_wmma_f32_16x16x32_bf16(false, a, false, b, (short)0, c, false, false);
}

// ---------------------------------------------------------------------------
// Dense: Y[rows,128] = act(X[rows,128] @ W[128,128] + b); ACT: 0=none, 1=tanh(ssp(.))
// One wave handles a 16-row tile. Weights transposed to bf16 in LDS.
// ---------------------------------------------------------------------------
template <int ACT>
__global__ __launch_bounds__(256) void dense_kernel(
    const float* __restrict__ X, const float* __restrict__ W,
    const float* __restrict__ bias, float* __restrict__ Y, int nrows)
{
    __shared__ __align__(32) __bf16 sWT[HID * HID];   // [n][k]
    __shared__ float sB[HID];
    int t = threadIdx.x;
    for (int i = t; i < HID * HID; i += 256) {
        int n = i >> 7, k = i & 127;
        sWT[n * HID + k] = (__bf16)W[k * HID + n];
    }
    if (t < HID) sB[t] = bias[t];
    __syncthreads();

    int lane = t & 31;
    int half = lane >> 4;
    int lrow = lane & 15;
    int wave = (blockIdx.x * 256 + t) >> 5;
    int ntile = nrows >> 4;
    if (wave >= ntile) return;

    const float* xp = X + (size_t)(wave * 16 + lrow) * HID;
    bf16x16 a[4];
#pragma unroll
    for (int c = 0; c < 4; ++c) a[c] = pack_a_f32(xp, c * 32 + half * 8);

#pragma unroll
    for (int j = 0; j < 8; ++j) {
        int col = j * 16 + lrow;
        float bv = sB[col];
        f32x8 acc;
#pragma unroll
        for (int r = 0; r < 8; ++r) acc[r] = bv;
#pragma unroll
        for (int c = 0; c < 4; ++c) {
            const __bf16* bp = &sWT[col * HID + c * 32 + half * 8];
            bf16x8 blo = *(const bf16x8*)bp, bhi = *(const bf16x8*)(bp + 16);
            bf16x16 b;
#pragma unroll
            for (int i = 0; i < 8; ++i) { b[i] = blo[i]; b[8 + i] = bhi[i]; }
            acc = wmma_bf16(a[c], b, acc);
        }
        float* yp = Y + (size_t)(wave * 16) * HID + col;
#pragma unroll
        for (int r = 0; r < 8; ++r) {
            float v = acc[r];
            if (ACT == 1) v = tanhf(ssp(v));
            yp[(size_t)(r + half * 8) * HID] = v;
        }
    }
}

// ---------------------------------------------------------------------------
// Fused edge kernel: per 32-edge tile (one wave, two 16-row M-tiles sharing
// every B operand):
//   T  = ssp(E[32x64] @ W1 + b1)           (GEMM1, bf16 WMMA, staged to LDS)
//   he = ssp(T @ W2 + b2)                  (GEMM2, bf16 WMMA)
//   atomicAdd(agg[dst], hv[src] * he)      (gather + filter + scatter)
// ---------------------------------------------------------------------------
__global__ __launch_bounds__(256) void edge_kernel(
    const float* __restrict__ E, const int* __restrict__ src, const int* __restrict__ dst,
    const float* __restrict__ hv, float* __restrict__ agg,
    const float* __restrict__ W1, const float* __restrict__ b1,
    const float* __restrict__ W2, const float* __restrict__ b2,
    int ntiles, int totalWaves)
{
    __shared__ __align__(32) __bf16 sW1T[HID * EIN];    // [n][k], k<64  (16 KB)
    __shared__ __align__(32) __bf16 sW2T[HID * HID];    // [n][k]        (32 KB)
    __shared__ __align__(32) __bf16 sT[8 * 32 * HID];   // per-wave 32x128 (64 KB)
    __shared__ float sB1[HID], sB2[HID];

    int t = threadIdx.x;
    for (int i = t; i < HID * EIN; i += 256) {
        int n = i >> 6, k = i & 63;
        sW1T[n * EIN + k] = (__bf16)W1[k * HID + n];
    }
    for (int i = t; i < HID * HID; i += 256) {
        int n = i >> 7, k = i & 127;
        sW2T[n * HID + k] = (__bf16)W2[k * HID + n];
    }
    if (t < HID) { sB1[t] = b1[t]; sB2[t] = b2[t]; }
    __syncthreads();

    int lane = t & 31;
    int half = lane >> 4;
    int lrow = lane & 15;
    int wid  = t >> 5;
    int gwave = blockIdx.x * 8 + wid;
    __bf16* myT = &sT[wid * 32 * HID];

    for (int tile = gwave; tile < ntiles; tile += totalWaves) {
        int base = tile * 32;
        int sidx[2][8], didx[2][8];
#pragma unroll
        for (int p = 0; p < 2; ++p)
#pragma unroll
            for (int r = 0; r < 8; ++r) {
                int m = base + p * 16 + r + half * 8;
                sidx[p][r] = src[m];
                didx[p][r] = dst[m];
            }

        // --- GEMM1: T = ssp(E @ W1 + b1), bf16 T to LDS (row-major 32x128)
        bf16x16 a1[2][2];
#pragma unroll
        for (int p = 0; p < 2; ++p) {
            const float* ep = E + (size_t)(base + p * 16 + lrow) * EIN;
#pragma unroll
            for (int c = 0; c < 2; ++c) a1[p][c] = pack_a_f32(ep, c * 32 + half * 8);
        }

#pragma unroll
        for (int j = 0; j < 8; ++j) {
            int col = j * 16 + lrow;
            float bv = sB1[col];
            f32x8 acc0, acc1;
#pragma unroll
            for (int r = 0; r < 8; ++r) { acc0[r] = bv; acc1[r] = bv; }
#pragma unroll
            for (int c = 0; c < 2; ++c) {
                const __bf16* bp = &sW1T[col * EIN + c * 32 + half * 8];
                bf16x8 blo = *(const bf16x8*)bp, bhi = *(const bf16x8*)(bp + 16);
                bf16x16 b;
#pragma unroll
                for (int i = 0; i < 8; ++i) { b[i] = blo[i]; b[8 + i] = bhi[i]; }
                acc0 = wmma_bf16(a1[0][c], b, acc0);
                acc1 = wmma_bf16(a1[1][c], b, acc1);
            }
#pragma unroll
            for (int r = 0; r < 8; ++r) {
                myT[(r + half * 8) * HID + col]      = (__bf16)ssp(acc0[r]);
                myT[(16 + r + half * 8) * HID + col] = (__bf16)ssp(acc1[r]);
            }
        }
        asm volatile("s_wait_dscnt 0" ::: "memory");

        // --- Reload T as A-operands for GEMM2
        bf16x16 a2[2][4];
#pragma unroll
        for (int p = 0; p < 2; ++p)
#pragma unroll
            for (int c = 0; c < 4; ++c) {
                int kb = c * 32 + half * 8;
                const __bf16* tp = &myT[(p * 16 + lrow) * HID + kb];
                bf16x8 lo = *(const bf16x8*)tp;
                bf16x8 hi = *(const bf16x8*)(tp + 16);
                bf16x16 v;
#pragma unroll
                for (int i = 0; i < 8; ++i) { v[i] = lo[i]; v[8 + i] = hi[i]; }
                a2[p][c] = v;
            }

        // --- GEMM2 + ssp + gather(hv[src]) * he -> atomic scatter into agg[dst]
#pragma unroll
        for (int j = 0; j < 8; ++j) {
            int col = j * 16 + lrow;
            float bv = sB2[col];
            f32x8 acc0, acc1;
#pragma unroll
            for (int r = 0; r < 8; ++r) { acc0[r] = bv; acc1[r] = bv; }
#pragma unroll
            for (int c = 0; c < 4; ++c) {
                const __bf16* bp = &sW2T[col * HID + c * 32 + half * 8];
                bf16x8 blo = *(const bf16x8*)bp, bhi = *(const bf16x8*)(bp + 16);
                bf16x16 b;
#pragma unroll
                for (int i = 0; i < 8; ++i) { b[i] = blo[i]; b[8 + i] = bhi[i]; }
                acc0 = wmma_bf16(a2[0][c], b, acc0);
                acc1 = wmma_bf16(a2[1][c], b, acc1);
            }
#pragma unroll
            for (int r = 0; r < 8; ++r) {
                float he0 = ssp(acc0[r]);
                float he1 = ssp(acc1[r]);
                float x0 = hv[(size_t)sidx[0][r] * HID + col];
                float x1 = hv[(size_t)sidx[1][r] * HID + col];
                atomAddF(&agg[(size_t)didx[0][r] * HID + col], he0 * x0);
                atomAddF(&agg[(size_t)didx[1][r] * HID + col], he1 * x1);
            }
        }
    }
}

__global__ __launch_bounds__(256) void zero_kernel(float4* __restrict__ p, int n4) {
    int i = blockIdx.x * 256 + threadIdx.x;
    if (i < n4) p[i] = make_float4(0.f, 0.f, 0.f, 0.f);
}

extern "C" void kernel_launch(void* const* d_in, const int* in_sizes, int n_in,
                              void* d_out, int out_size, void* d_ws, size_t ws_size,
                              hipStream_t stream) {
    const float* node = (const float*)d_in[0];
    const float* edge = (const float*)d_in[1];
    const int*   src  = (const int*)d_in[2];
    const int*   dst  = (const int*)d_in[3];
    const float* n1W = (const float*)d_in[4],  *n1b = (const float*)d_in[5];
    const float* e1W1= (const float*)d_in[6],  *e1b1= (const float*)d_in[7];
    const float* e1W2= (const float*)d_in[8],  *e1b2= (const float*)d_in[9];
    const float* o1W = (const float*)d_in[10], *o1b = (const float*)d_in[11];
    const float* n2W = (const float*)d_in[12], *n2b = (const float*)d_in[13];
    const float* e2W1= (const float*)d_in[14], *e2b1= (const float*)d_in[15];
    const float* e2W2= (const float*)d_in[16], *e2b2= (const float*)d_in[17];
    const float* o2W = (const float*)d_in[18], *o2b = (const float*)d_in[19];
    float* out  = (float*)d_out;
    float* buf0 = (float*)d_ws;
    float* buf1 = buf0 + (size_t)NNODES * HID;

    const int denseBlocks = ((NNODES / 16) + 7) / 8;           // 157
    const int n4 = NNODES * HID / 4;
    const int zeroBlocks = (n4 + 255) / 256;
    const int edgeBlocks = 1250;
    const int totalWaves = edgeBlocks * 8;
    const int etiles = NEDGES / 32;                            // 20000

    // conv1
    dense_kernel<0><<<denseBlocks, 256, 0, stream>>>(node, n1W, n1b, buf0, NNODES); // hv1
    zero_kernel<<<zeroBlocks, 256, 0, stream>>>((float4*)buf1, n4);                 // agg1 = 0
    edge_kernel<<<edgeBlocks, 256, 0, stream>>>(edge, src, dst, buf0, buf1,
                                                e1W1, e1b1, e1W2, e1b2, etiles, totalWaves);
    dense_kernel<1><<<denseBlocks, 256, 0, stream>>>(buf1, o1W, o1b, buf0, NNODES); // h1 = tanh(ssp(.))
    // conv2
    dense_kernel<0><<<denseBlocks, 256, 0, stream>>>(buf0, n2W, n2b, buf1, NNODES); // hv2
    zero_kernel<<<zeroBlocks, 256, 0, stream>>>((float4*)buf0, n4);                 // agg2 = 0
    edge_kernel<<<edgeBlocks, 256, 0, stream>>>(edge, src, dst, buf1, buf0,
                                                e2W1, e2b1, e2W2, e2b2, etiles, totalWaves);
    dense_kernel<1><<<denseBlocks, 256, 0, stream>>>(buf0, o2W, o2b, out, NNODES);  // tanh(ssp(.))
}
